// HetMoE_64819646431483
// MI455X (gfx1250) — matrix operand order repaired and checked by reading, not compiled, not run on hardware
//
#include <hip/hip_runtime.h>
#include <hip/hip_bf16.h>
#include <math.h>

#define DIM   768
#define NTOK  4096
#define NEXP  8
#define CAP   NTOK
#define HID   3072
#define MIDW  1536

#define BM 128
#define BN 128
#define BK 32
#define LDSW 20          // u32 per LDS row: 16 data + 4 pad (16B-aligned rows, no bank conflicts)
#define BUFU ((BM + BN) * LDSW)

typedef __attribute__((ext_vector_type(16))) __bf16       bf16x16;
typedef __attribute__((ext_vector_type(8)))  float        f32x8;
typedef __attribute__((ext_vector_type(8)))  unsigned int u32x8;

__device__ __forceinline__ unsigned short f2bf(float f) {
  union { float f; unsigned u; } v; v.f = f;
  unsigned r = v.u + 0x7FFFu + ((v.u >> 16) & 1u);   // round-to-nearest-even
  return (unsigned short)(r >> 16);
}
__device__ __forceinline__ float bf2f(unsigned short b) {
  union { unsigned u; float f; } v; v.u = ((unsigned)b) << 16;
  return v.f;
}
__device__ __forceinline__ float gelu_f(float z) {
  return 0.5f * z * (1.0f + erff(z * 0.70710678118654752f));
}
__device__ __forceinline__ float silu_f(float z) {
  return z / (1.0f + __expf(-z));
}

// CDNA5 async global->LDS copy (ASYNCcnt). LDS byte address = low 32 bits of
// the generic pointer (addrspace(3) offset on AMDGPU).
__device__ __forceinline__ void async_b128(unsigned ldsByteAddr, const void* g) {
  asm volatile("global_load_async_to_lds_b128 %0, %1, off"
               :: "v"(ldsByteAddr), "v"(g) : "memory");
}
__device__ __forceinline__ void wait_async0() {
  asm volatile("s_wait_asynccnt 0" ::: "memory");
}

// ---------------------------------------------------------------------------
// Shared GEMM helpers: fragment gathers (documented CDNA5 operand layouts) and
// the 2x4 subtile WMMA block; epilogue with activation / scatter-accumulate.
// ---------------------------------------------------------------------------
__device__ __forceinline__ void tiles_wmma(const unsigned int* __restrict__ bufA,
                                           const unsigned int* __restrict__ bufB,
                                           int lane, int waveRow, int waveCol,
                                           f32x8 (&acc)[2][4])
{
  const int half = lane >> 4;
  const int mn   = lane & 15;

  bf16x16 aF[2];
#pragma unroll
  for (int sm = 0; sm < 2; ++sm) {
    int r = waveRow * 32 + sm * 16 + mn;
    u32x8 u;
#pragma unroll
    for (int pq = 0; pq < 8; ++pq) {
      int kp = ((pq >> 2) * 8) + half * 4 + (pq & 3);   // A-operand interleave
      u[pq] = bufA[r * LDSW + kp];
    }
    aF[sm] = __builtin_bit_cast(bf16x16, u);
  }
  bf16x16 bF[4];
#pragma unroll
  for (int sn = 0; sn < 4; ++sn) {
    int r = waveCol * 64 + sn * 16 + mn;
    u32x8 u;
#pragma unroll
    for (int pq = 0; pq < 8; ++pq) {
      int kp = ((pq >> 2) * 8) + half * 4 + (pq & 3);   // B-operand interleave (same as A)
      u[pq] = bufB[r * LDSW + kp];
    }
    bF[sn] = __builtin_bit_cast(bf16x16, u);
  }
#pragma unroll
  for (int sm = 0; sm < 2; ++sm)
#pragma unroll
    for (int sn = 0; sn < 4; ++sn)
      acc[sm][sn] = __builtin_amdgcn_wmma_f32_16x16x32_bf16(
          false, aF[sm], false, bF[sn], (short)0, acc[sm][sn], false, false);
}

__device__ __forceinline__ void epilogue(f32x8 (&acc)[2][4], int cnt,
                                         int rowBase, int colBase,
                                         int lane, int waveRow, int waveCol,
                                         const int* __restrict__ toks,
                                         const float* __restrict__ wts,
                                         unsigned short* __restrict__ dstB,
                                         float* __restrict__ outF,
                                         int Nout, int act, int mode)
{
  const int half = lane >> 4;
  const int nloc = lane & 15;
#pragma unroll
  for (int sm = 0; sm < 2; ++sm) {
#pragma unroll
    for (int j = 0; j < 8; ++j) {
      int prow = rowBase + waveRow * 32 + sm * 16 + half * 8 + j;
      if (prow >= cnt) continue;
#pragma unroll
      for (int sn = 0; sn < 4; ++sn) {
        int col = colBase + waveCol * 64 + sn * 16 + nloc;
        float v = acc[sm][sn][j];
        if (mode == 0) {
          if (act == 1)      v = gelu_f(v);
          else if (act == 2) v = silu_f(v);
          dstB[(size_t)prow * Nout + col] = f2bf(v);
        } else {
          int t = toks[prow];
          outF[(size_t)t * DIM + col] += wts[prow] * v;  // experts serialized on stream
        }
      }
    }
  }
}

// ---------------------------------------------------------------------------
// Gate: logits -> softmax -> top-2 routing lists + bf16 conversion of x
// ---------------------------------------------------------------------------
__global__ void gate_kernel(const float* __restrict__ x,
                            const float* __restrict__ gw,      // [8,768]
                            int* __restrict__ cnt, int* __restrict__ freqc,
                            float* __restrict__ probsum,
                            int* __restrict__ lists, float* __restrict__ wgts,
                            unsigned short* __restrict__ xb)
{
  const int lane = threadIdx.x & 31;
  const int wid  = threadIdx.x >> 5;
  const int tok  = blockIdx.x * 8 + wid;
  if (tok >= NTOK) return;

  float l[NEXP];
#pragma unroll
  for (int e = 0; e < NEXP; ++e) l[e] = 0.0f;

  for (int j = lane; j < DIM; j += 32) {
    float xv = x[(size_t)tok * DIM + j];
    xb[(size_t)tok * DIM + j] = f2bf(xv);
#pragma unroll
    for (int e = 0; e < NEXP; ++e) l[e] += xv * gw[e * DIM + j];
  }
#pragma unroll
  for (int e = 0; e < NEXP; ++e) {
    float s = l[e];
    for (int off = 16; off > 0; off >>= 1) s += __shfl_xor(s, off, 32);
    l[e] = s;
  }
  if (lane == 0) {
    const float FMAX = 3.4028235e38f;
    float m = -FMAX;
#pragma unroll
    for (int e = 0; e < NEXP; ++e) {
      float z = l[e];
      if (z != z) z = 0.0f;               // nan_to_num
      z = fminf(fmaxf(z, -FMAX), FMAX);
      l[e] = z;
      m = fmaxf(m, z);
    }
    float sum = 0.0f, pr[NEXP];
#pragma unroll
    for (int e = 0; e < NEXP; ++e) { pr[e] = __expf(l[e] - m); sum += pr[e]; }
#pragma unroll
    for (int e = 0; e < NEXP; ++e) { pr[e] /= sum; atomicAdd(&probsum[e], pr[e]); }
    int i0 = 0; float v0 = pr[0];
#pragma unroll
    for (int e = 1; e < NEXP; ++e) if (pr[e] > v0) { v0 = pr[e]; i0 = e; }
    int i1 = (i0 == 0) ? 1 : 0; float v1 = pr[i1];
#pragma unroll
    for (int e = 0; e < NEXP; ++e) {
      if (e == i0 || e == i1) continue;
      if (pr[e] > v1) { v1 = pr[e]; i1 = e; }
    }
    float inv = 1.0f / (v0 + v1 + 1e-8f);
    atomicAdd(&freqc[i0], 1);
    atomicAdd(&freqc[i1], 1);
    int p0 = atomicAdd(&cnt[i0], 1);
    lists[i0 * CAP + p0] = tok; wgts[i0 * CAP + p0] = v0 * inv;
    int p1 = atomicAdd(&cnt[i1], 1);
    lists[i1 * CAP + p1] = tok; wgts[i1 * CAP + p1] = v1 * inv;
  }
}

// ---------------------------------------------------------------------------
// Preferred GEMM: bf16 A and bf16 W, async double-buffered LDS staging.
// ---------------------------------------------------------------------------
__global__ __launch_bounds__(256)
void moe_gemm_bf16(const unsigned short* __restrict__ Wb,   // [Nout,K] bf16
                   const unsigned short* __restrict__ A,    // [*,K] bf16
                   const int* __restrict__ gatherRows,
                   const int* __restrict__ toks,
                   const float* __restrict__ wts,
                   const int* __restrict__ cntp,
                   unsigned short* __restrict__ dstB,
                   float* __restrict__ outF,
                   int K, int Nout, int act, int mode)
{
  const int cnt = cntp[0];
  const int rowBase = blockIdx.x * BM;
  if (rowBase >= cnt) return;
  const int colBase = blockIdx.y * BN;

  __shared__ unsigned int lds[2 * BUFU];

  const int tid     = threadIdx.x;
  const int lane    = tid & 31;
  const int wid     = tid >> 5;
  const int waveRow = wid >> 1;
  const int waveCol = wid & 1;

  const int lr   = tid >> 1;          // local row 0..127
  const int kSel = tid & 1;           // which 16-element half of BK

  int p  = rowBase + lr;
  int pc = p < cnt ? p : (cnt - 1);
  int srcRow = gatherRows ? gatherRows[pc] : pc;
  const unsigned short* aRow = A  + (size_t)srcRow * K + kSel * 16;
  const unsigned short* wRow = Wb + (size_t)(colBase + lr) * K + kSel * 16;

  const unsigned lds0 = (unsigned)(unsigned long long)&lds[0];
  const unsigned aOff = (unsigned)((lr * LDSW + kSel * 8) * 4);
  const unsigned bOff = (unsigned)((BM * LDSW + lr * LDSW + kSel * 8) * 4);

  auto issue = [&](int kt, int buf) {
    const unsigned short* a0 = aRow + (size_t)kt * BK;
    const unsigned short* w0 = wRow + (size_t)kt * BK;
    unsigned lbase = lds0 + (unsigned)buf * (BUFU * 4u);
    async_b128(lbase + aOff,      a0);
    async_b128(lbase + aOff + 16, a0 + 8);
    async_b128(lbase + bOff,      w0);
    async_b128(lbase + bOff + 16, w0 + 8);
  };

  f32x8 acc[2][4];
#pragma unroll
  for (int i = 0; i < 2; ++i)
#pragma unroll
    for (int j = 0; j < 4; ++j)
      acc[i][j] = (f32x8){0.f,0.f,0.f,0.f,0.f,0.f,0.f,0.f};

  const int nK = K / BK;
  issue(0, 0);
  for (int kt = 0; kt < nK; ++kt) {
    wait_async0();         // my tile-kt transfers resident
    __syncthreads();       // everyone's resident; everyone done reading other buffer
    if (kt + 1 < nK) issue(kt + 1, (kt + 1) & 1);
    const unsigned int* bufA = &lds[(kt & 1) * BUFU];
    tiles_wmma(bufA, bufA + BM * LDSW, lane, waveRow, waveCol, acc);
  }
  epilogue(acc, cnt, rowBase, colBase, lane, waveRow, waveCol,
           toks, wts, dstB, outF, Nout, act, mode);
}

// ---------------------------------------------------------------------------
// Fallback GEMM (small-workspace): W in fp32, converted while staging.
// ---------------------------------------------------------------------------
__global__ __launch_bounds__(256)
void moe_gemm_f32(const float* __restrict__ W,
                  const unsigned short* __restrict__ A,
                  const int* __restrict__ gatherRows,
                  const int* __restrict__ toks,
                  const float* __restrict__ wts,
                  const int* __restrict__ cntp,
                  unsigned short* __restrict__ dstB,
                  float* __restrict__ outF,
                  int K, int Nout, int act, int mode)
{
  const int cnt = cntp[0];
  const int rowBase = blockIdx.x * BM;
  if (rowBase >= cnt) return;
  const int colBase = blockIdx.y * BN;

  __shared__ unsigned int lds[BUFU];

  const int tid     = threadIdx.x;
  const int lane    = tid & 31;
  const int wid     = tid >> 5;
  const int waveRow = wid >> 1;
  const int waveCol = wid & 1;

  const int lr   = tid >> 1;
  const int kSel = tid & 1;

  int p  = rowBase + lr;
  int pc = p < cnt ? p : (cnt - 1);
  int srcRow = gatherRows ? gatherRows[pc] : pc;
  const unsigned short* aRow = A + (size_t)srcRow * K + kSel * 16;
  const float*          wRow = W + (size_t)(colBase + lr) * K + kSel * 16;

  f32x8 acc[2][4];
#pragma unroll
  for (int i = 0; i < 2; ++i)
#pragma unroll
    for (int j = 0; j < 4; ++j)
      acc[i][j] = (f32x8){0.f,0.f,0.f,0.f,0.f,0.f,0.f,0.f};

  for (int k0 = 0; k0 < K; k0 += BK) {
    {
      const uint4* src = (const uint4*)(aRow + k0);
      uint4 d0 = src[0], d1 = src[1];
      unsigned int* dst = &lds[lr * LDSW + kSel * 8];
      dst[0]=d0.x; dst[1]=d0.y; dst[2]=d0.z; dst[3]=d0.w;
      dst[4]=d1.x; dst[5]=d1.y; dst[6]=d1.z; dst[7]=d1.w;
    }
    {
      const float4* src = (const float4*)(wRow + k0);
      float4 f0 = src[0], f1 = src[1], f2 = src[2], f3 = src[3];
      unsigned int* dst = &lds[BM * LDSW + lr * LDSW + kSel * 8];
      dst[0] = ((unsigned)f2bf(f0.y) << 16) | f2bf(f0.x);
      dst[1] = ((unsigned)f2bf(f0.w) << 16) | f2bf(f0.z);
      dst[2] = ((unsigned)f2bf(f1.y) << 16) | f2bf(f1.x);
      dst[3] = ((unsigned)f2bf(f1.w) << 16) | f2bf(f1.z);
      dst[4] = ((unsigned)f2bf(f2.y) << 16) | f2bf(f2.x);
      dst[5] = ((unsigned)f2bf(f2.w) << 16) | f2bf(f2.z);
      dst[6] = ((unsigned)f2bf(f3.y) << 16) | f2bf(f3.x);
      dst[7] = ((unsigned)f2bf(f3.w) << 16) | f2bf(f3.z);
    }
    __syncthreads();
    tiles_wmma(lds, lds + BM * LDSW, lane, waveRow, waveCol, acc);
    __syncthreads();
  }
  epilogue(acc, cnt, rowBase, colBase, lane, waveRow, waveCol,
           toks, wts, dstB, outF, Nout, act, mode);
}

// fp32 -> bf16 bulk conversion (weights), 4 elements/thread
__global__ void cvt_bf16_kernel(const float4* __restrict__ s,
                                uint2* __restrict__ d, int n4)
{
  int i = blockIdx.x * 256 + threadIdx.x;
  if (i < n4) {
    float4 f = s[i];
    uint2 o;
    o.x = ((unsigned)f2bf(f.y) << 16) | f2bf(f.x);
    o.y = ((unsigned)f2bf(f.w) << 16) | f2bf(f.z);
    d[i] = o;
  }
}

// SwiGLU elementwise: h1 = silu(xW1) * (xW2)
__global__ void swiglu_mul(unsigned short* __restrict__ h1,
                           const unsigned short* __restrict__ h2,
                           const int* __restrict__ cntp, int H)
{
  long long i = (long long)blockIdx.x * blockDim.x + threadIdx.x;
  long long n = (long long)cntp[0] * H;
  if (i < n) h1[i] = f2bf(bf2f(h1[i]) * bf2f(h2[i]));
}

__global__ void aux_kernel(const float* __restrict__ probsum,
                           const int* __restrict__ freqc,
                           float* __restrict__ outAux)
{
  float s = 0.0f;
  for (int e = 0; e < NEXP; ++e)
    s += (probsum[e] / (float)NTOK) * ((float)freqc[e] / (float)(NTOK * 2));
  outAux[0] = 0.01f * (float)NEXP * s;
}

// ---------------------------------------------------------------------------
extern "C" void kernel_launch(void* const* d_in, const int* in_sizes, int n_in,
                              void* d_out, int out_size, void* d_ws, size_t ws_size,
                              hipStream_t stream)
{
  (void)in_sizes; (void)n_in;
  const float* x  = (const float*)d_in[0];
  const float* gw = (const float*)d_in[1];

  static const int tshape[4][4][2] = {
    {{HID,DIM},{DIM,HID},{0,0},{0,0}},            // StandardFFN
    {{HID,DIM},{HID,DIM},{DIM,HID},{0,0}},        // SwiGLU
    {{MIDW,DIM},{HID,MIDW},{MIDW,HID},{DIM,MIDW}},// DeepNarrow
    {{DIM,DIM},{HID,DIM},{DIM,HID},{0,0}}         // Bottleneck
  };
  static const int tnum[4] = {2, 3, 4, 3};

  const float* ew[NEXP][4];
  int idx = 2;
  for (int e = 0; e < NEXP; ++e)
    for (int k = 0; k < tnum[e % 4]; ++k) ew[e][k] = (const float*)d_in[idx++];

  // workspace layout
  char*  ws      = (char*)d_ws;
  int*   cnt     = (int*)  (ws + 0);
  int*   freqc   = (int*)  (ws + 32);
  float* probsum = (float*)(ws + 64);
  int*   lists   = (int*)  (ws + 256);
  float* wgts    = (float*)(ws + 256 + (size_t)NEXP * CAP * 4);
  unsigned short* xb = (unsigned short*)(ws + 256 + 2 * (size_t)NEXP * CAP * 4);
  size_t off = 256 + 2 * (size_t)NEXP * CAP * 4 + (size_t)NTOK * DIM * 2;
  off = (off + 255) & ~(size_t)255;
  unsigned short* h1 = (unsigned short*)(ws + off);
  off += (size_t)NTOK * HID * 2;
  unsigned short* h2 = (unsigned short*)(ws + off);
  off += (size_t)NTOK * HID * 2;
  off = (off + 255) & ~(size_t)255;

  // bf16 weight cache region
  unsigned short* wb[NEXP][4];
  size_t woff = off;
  for (int e = 0; e < NEXP; ++e)
    for (int k = 0; k < tnum[e % 4]; ++k) {
      wb[e][k] = (unsigned short*)(ws + woff);
      woff += (size_t)tshape[e % 4][k][0] * tshape[e % 4][k][1] * 2;
    }
  const bool useBf16 = (woff <= ws_size);

  float* out = (float*)d_out;

  hipMemsetAsync(d_out, 0, (size_t)out_size * sizeof(float), stream);
  hipMemsetAsync(d_ws, 0, 256, stream);

  if (useBf16) {
    for (int e = 0; e < NEXP; ++e)
      for (int k = 0; k < tnum[e % 4]; ++k) {
        int n4 = tshape[e % 4][k][0] * tshape[e % 4][k][1] / 4;
        cvt_bf16_kernel<<<(n4 + 255) / 256, 256, 0, stream>>>(
            (const float4*)ew[e][k], (uint2*)wb[e][k], n4);
      }
  }

  gate_kernel<<<NTOK / 8, 256, 0, stream>>>(x, gw, cnt, freqc, probsum,
                                            lists, wgts, xb);

  auto gemm = [&](int e, int k, const unsigned short* A, const int* gather,
                  const int* tk, const float* wt, const int* c,
                  unsigned short* dB, int K, int Nout, int act, int mode) {
    dim3 grid(CAP / BM, Nout / BN);
    if (useBf16)
      moe_gemm_bf16<<<grid, 256, 0, stream>>>(wb[e][k], A, gather, tk, wt, c,
                                              dB, out, K, Nout, act, mode);
    else
      moe_gemm_f32<<<grid, 256, 0, stream>>>(ew[e][k], A, gather, tk, wt, c,
                                             dB, out, K, Nout, act, mode);
  };

  for (int e = 0; e < NEXP; ++e) {
    const int*   lst = lists + e * CAP;
    const float* wg  = wgts + e * CAP;
    const int*   c   = cnt + e;
    int t = e % 4;
    if (t == 0) {          // gelu(x@up.T)@down.T
      gemm(e, 0, xb, lst, lst, wg, c, h1, DIM, HID, 1, 0);
      gemm(e, 1, h1, nullptr, lst, wg, c, nullptr, HID, DIM, 0, 1);
    } else if (t == 1) {   // SwiGLU
      gemm(e, 0, xb, lst, lst, wg, c, h1, DIM, HID, 2, 0);
      gemm(e, 1, xb, lst, lst, wg, c, h2, DIM, HID, 0, 0);
      swiglu_mul<<<(NTOK * HID) / 256, 256, 0, stream>>>(h1, h2, c, HID);
      gemm(e, 2, h1, nullptr, lst, wg, c, nullptr, HID, DIM, 0, 1);
    } else if (t == 2) {   // DeepNarrow
      gemm(e, 0, xb, lst, lst, wg, c, h1, DIM, MIDW, 1, 0);
      gemm(e, 1, h1, nullptr, lst, wg, c, h2, MIDW, HID, 1, 0);
      gemm(e, 2, h2, nullptr, lst, wg, c, h1, HID, MIDW, 1, 0);
      gemm(e, 3, h1, nullptr, lst, wg, c, nullptr, MIDW, DIM, 0, 1);
    } else {               // Bottleneck
      gemm(e, 0, xb, lst, lst, wg, c, h1, DIM, DIM, 1, 0);
      gemm(e, 1, h1, nullptr, lst, wg, c, h2, DIM, HID, 1, 0);
      gemm(e, 2, h2, nullptr, lst, wg, c, nullptr, HID, DIM, 0, 1);
    }
  }
  aux_kernel<<<1, 1, 0, stream>>>(probsum, freqc, out + (size_t)NTOK * DIM);
}
